// MultiHeadAttention_90993177133622
// MI455X (gfx1250) — matrix operand order repaired and checked by reading, not compiled, not run on hardware
//
#include <hip/hip_runtime.h>

typedef unsigned short u16;
typedef unsigned int   u32;
typedef unsigned long long u64;
typedef __bf16 bf16_t;
typedef bf16_t v16bf __attribute__((ext_vector_type(16)));
typedef float  v8f   __attribute__((ext_vector_type(8)));
typedef u32    v4u   __attribute__((ext_vector_type(4)));
typedef float  v4f   __attribute__((ext_vector_type(4)));

union FragBF { v16bf v; v4u q[2]; };

__device__ __forceinline__ u16 f32_to_bf16(float f) {
  u32 u = __float_as_uint(f);
  u32 r = u + 0x7FFFu + ((u >> 16) & 1u);   // round-to-nearest-even
  return (u16)(r >> 16);
}
__device__ __forceinline__ float bf_lo(u32 w) { return __uint_as_float(w << 16); }
__device__ __forceinline__ float bf_hi(u32 w) { return __uint_as_float(w & 0xFFFF0000u); }

// ---------------------------------------------------------------------------
// f32 -> bf16 conversion, 4 elements / thread
// ---------------------------------------------------------------------------
__global__ void cvt_f32_to_bf16(const float* __restrict__ src,
                                u16* __restrict__ dst, int n4) {
  int i = blockIdx.x * blockDim.x + threadIdx.x;
  if (i >= n4) return;
  v4f x = ((const v4f*)src)[i];
  u32 lo = (u32)f32_to_bf16(x[0]) | ((u32)f32_to_bf16(x[1]) << 16);
  u32 hi = (u32)f32_to_bf16(x[2]) | ((u32)f32_to_bf16(x[3]) << 16);
  ((u32*)dst)[2 * i]     = lo;
  ((u32*)dst)[2 * i + 1] = hi;
}

// ---------------------------------------------------------------------------
// C[M,N] = A[M,K] @ W[N,K]^T + bias   (bf16 in, f32 accumulate via WMMA)
// block = 256 threads (8 waves, 2Mx4N), block tile 128x256, wave tile 64x64
// 16 WMMA per K-step (K step 32) vs 16 b128 loads -> ~32 FLOP/byte from cache
// ---------------------------------------------------------------------------
template <typename OutT>
__global__ __launch_bounds__(256, 1)
void gemm_bf16_wmma(const u16* __restrict__ A, const u16* __restrict__ Wt,
                    const float* __restrict__ bias, OutT* __restrict__ C,
                    int M, int N, int K) {
  const int lane = threadIdx.x & 31;
  const int wave = threadIdx.x >> 5;
  const int half = lane >> 4;   // K sub-chunk (A), K half (B), M half (C/D)
  const int l16  = lane & 15;   // M row (A) / N col (B, C)

  const int waveM = wave & 1;   // 2 waves along M
  const int waveN = wave >> 1;  // 4 waves along N
  const int mBase = blockIdx.y * 128 + waveM * 64;
  const int nBase = blockIdx.x * 256 + waveN * 64;

  v8f acc[4][4];
  for (int mt = 0; mt < 4; ++mt)
    for (int nt = 0; nt < 4; ++nt)
      for (int j = 0; j < 8; ++j) acc[mt][nt][j] = 0.0f;

  // Per-lane base pointers matching the ISA 16-bit WMMA operand layouts:
  //  A 16x32: lane(l16)=M; VGPR0-3 hold K = half*8..+7, VGPR4-7 hold K = 16+half*8..+7
  //  B 32x16: lane(l16)=N; VGPR0-3 hold K = half*8..+7, VGPR4-7 hold K = 16+half*8..+7
  const u16* aRow[4];
  for (int mt = 0; mt < 4; ++mt)
    aRow[mt] = A + (size_t)(mBase + mt * 16 + l16) * K + half * 8;
  const u16* bRow[4];
  for (int nt = 0; nt < 4; ++nt)
    bRow[nt] = Wt + (size_t)(nBase + nt * 16 + l16) * K + half * 8;

  for (int k0 = 0; k0 < K; k0 += 32) {
    FragBF a[4], b[4];
    for (int mt = 0; mt < 4; ++mt) {
      a[mt].q[0] = *(const v4u*)(aRow[mt] + k0);        // K = half*8 .. +7
      a[mt].q[1] = *(const v4u*)(aRow[mt] + k0 + 16);   // K = 16+half*8 .. +7
    }
    for (int nt = 0; nt < 4; ++nt) {
      b[nt].q[0] = *(const v4u*)(bRow[nt] + k0);        // K = half*8 .. +7
      b[nt].q[1] = *(const v4u*)(bRow[nt] + k0 + 16);   // K = 16+half*8 .. +7
    }
    if (k0 + 32 < K) {                                   // gfx1250 global_prefetch_b8
      __builtin_prefetch(aRow[0] + k0 + 32, 0, 3);
      __builtin_prefetch(bRow[0] + k0 + 32, 0, 3);
    }
    for (int mt = 0; mt < 4; ++mt)
      for (int nt = 0; nt < 4; ++nt)
        acc[mt][nt] = __builtin_amdgcn_wmma_f32_16x16x32_bf16(
            false, a[mt].v, false, b[nt].v, (short)0, acc[mt][nt], false, false);
  }

  // C/D layout: lanes 0-15 -> M=v, lanes 16-31 -> M=v+8; N = lane&15
  for (int mt = 0; mt < 4; ++mt)
    for (int nt = 0; nt < 4; ++nt) {
      const int col = nBase + nt * 16 + l16;
      const float bv = bias[col];
      for (int v = 0; v < 8; ++v) {
        const int row = mBase + mt * 16 + v + 8 * half;
        const float val = acc[mt][nt][v] + bv;
        if constexpr (sizeof(OutT) == 4) {
          ((float*)C)[(size_t)row * N + col] = val;
        } else {
          ((u16*)C)[(size_t)row * N + col] = f32_to_bf16(val);
        }
      }
    }
}

// ---------------------------------------------------------------------------
// Per-token head-axis attention: P = (Q Kt)*8, causal mask over [H,H],
// softmax, out = P V.   One wave per token; 4 waves / block.
// Q/K/V bf16 [T, H=16, D=64]; staged into LDS with async global->LDS copies.
// ---------------------------------------------------------------------------
__global__ __launch_bounds__(128, 4)
void attn_headmix(const u16* __restrict__ Qb, const u16* __restrict__ Kb,
                  const u16* __restrict__ Vb, u16* __restrict__ Ob,
                  const int* __restrict__ is_causal) {
  extern __shared__ u16 smem[];          // 4 waves * 3 tensors * 1024 bf16 = 24KB
  const int lane = threadIdx.x & 31;
  const int wave = threadIdx.x >> 5;
  const size_t tok  = (size_t)blockIdx.x * 4 + wave;
  const size_t base = tok * 1024;        // elements

  u16* sQ = smem + wave * 3072;
  u16* sK = sQ + 1024;
  u16* sV = sQ + 2048;

  // ---- async global -> LDS staging (gfx1250 GLOBAL_LOAD_ASYNC_TO_LDS_B128) ----
  // Dynamic LDS starts at offset 0 (no static __shared__ in this kernel), so the
  // LDS-address operand is the raw byte offset. Each lane moves 16B per chunk;
  // 4 chunks x 3 tensors = 2KB per tensor per wave.
  {
    const u32 ldsW = (u32)(wave * 6144 + lane * 16);     // bytes
    const u64 gq = (u64)(const char*)(Qb + base) + (u64)(lane * 16);
    const u64 gk = (u64)(const char*)(Kb + base) + (u64)(lane * 16);
    const u64 gv = (u64)(const char*)(Vb + base) + (u64)(lane * 16);
    for (int c = 0; c < 4; ++c) {
      const u32 lq = ldsW + (u32)(c * 512);
      const u32 lk = lq + 2048u;
      const u32 lv = lq + 4096u;
      asm volatile("global_load_async_to_lds_b128 %0, %1, off"
                   :: "v"(lq), "v"(gq + (u64)(c * 512)) : "memory");
      asm volatile("global_load_async_to_lds_b128 %0, %1, off"
                   :: "v"(lk), "v"(gk + (u64)(c * 512)) : "memory");
      asm volatile("global_load_async_to_lds_b128 %0, %1, off"
                   :: "v"(lv), "v"(gv + (u64)(c * 512)) : "memory");
    }
    asm volatile("s_wait_asynccnt 0x0" ::: "memory");    // each wave reads only
  }                                                       // its own staged data

  const int h  = lane & 15;   // output head row (lanes 0-15 / 16-31 mirror it)
  const int dh = lane >> 4;   // which 32-wide half of D this lane produces
  const bool causal = (is_causal[0] != 0);

  // preload Q row h as f32
  float qr[64];
  {
    const u32* qw = (const u32*)(sQ + h * 64);
    for (int d = 0; d < 32; ++d) {
      u32 a = qw[d];
      qr[2 * d]     = bf_lo(a);
      qr[2 * d + 1] = bf_hi(a);
    }
  }

  // P row h entirely in-lane -> in-lane softmax (no shuffles)
  float p[16];
  float mx = -3.0e38f;
  for (int g = 0; g < 16; ++g) {
    const u32* kw = (const u32*)(sK + g * 64);
    float s = 0.0f;
    for (int d = 0; d < 32; ++d) {
      u32 b = kw[d];
      s = fmaf(qr[2 * d],     bf_lo(b), s);
      s = fmaf(qr[2 * d + 1], bf_hi(b), s);
    }
    s *= 8.0f;                           // reference MULTIPLIES by sqrt(D)=8
    if (causal && g > h) s = -3.0e38f;
    p[g] = s;
    mx = fmaxf(mx, s);
  }
  float sum = 0.0f;
  for (int g = 0; g < 16; ++g) { p[g] = __expf(p[g] - mx); sum += p[g]; }
  const float inv = 1.0f / sum;

  float acc[32];
  for (int j = 0; j < 32; ++j) acc[j] = 0.0f;
  for (int g = 0; g < 16; ++g) {
    const float pg = p[g] * inv;
    const u32* vw = (const u32*)(sV + g * 64 + dh * 32);
    for (int j = 0; j < 16; ++j) {
      u32 b = vw[j];
      acc[2 * j]     = fmaf(pg, bf_lo(b), acc[2 * j]);
      acc[2 * j + 1] = fmaf(pg, bf_hi(b), acc[2 * j + 1]);
    }
  }

  u32* out = (u32*)(Ob + base + h * 64 + dh * 32);
  for (int j = 0; j < 16; ++j)
    out[j] = (u32)f32_to_bf16(acc[2 * j]) | ((u32)f32_to_bf16(acc[2 * j + 1]) << 16);
}

// ---------------------------------------------------------------------------
extern "C" void kernel_launch(void* const* d_in, const int* in_sizes, int n_in,
                              void* d_out, int out_size, void* d_ws, size_t ws_size,
                              hipStream_t stream) {
  (void)in_sizes; (void)n_in; (void)out_size; (void)ws_size;

  const float* QKV = (const float*)d_in[0];
  const float* Wq  = (const float*)d_in[1];
  const float* bq  = (const float*)d_in[2];
  const float* Wk  = (const float*)d_in[3];
  const float* bk  = (const float*)d_in[4];
  const float* Wv  = (const float*)d_in[5];
  const float* bv  = (const float*)d_in[6];
  const float* Wo  = (const float*)d_in[7];
  const float* bo  = (const float*)d_in[8];
  const int* is_causal = (const int*)d_in[9];

  const int M = 16384, N = 1024, K = 1024;           // M = B*S
  const size_t ME = (size_t)M * N;                    // 16.7M elems
  const size_t EE = (size_t)N * K;                    // 1M elems

  // workspace layout (bf16 / u16 elements), ~136 MB total
  u16* xb  = (u16*)d_ws;        // QKV bf16; later reused as attention output
  u16* wqb = xb  + ME;
  u16* wkb = wqb + EE;
  u16* wvb = wkb + EE;
  u16* wob = wvb + EE;
  u16* qb  = wob + EE;
  u16* kb  = qb  + ME;
  u16* vb  = kb  + ME;

  // 1) conversions
  cvt_f32_to_bf16<<<(int)(ME / 4 / 256), 256, 0, stream>>>(QKV, xb, (int)(ME / 4));
  cvt_f32_to_bf16<<<(int)(EE / 4 / 256), 256, 0, stream>>>(Wq, wqb, (int)(EE / 4));
  cvt_f32_to_bf16<<<(int)(EE / 4 / 256), 256, 0, stream>>>(Wk, wkb, (int)(EE / 4));
  cvt_f32_to_bf16<<<(int)(EE / 4 / 256), 256, 0, stream>>>(Wv, wvb, (int)(EE / 4));
  cvt_f32_to_bf16<<<(int)(EE / 4 / 256), 256, 0, stream>>>(Wo, wob, (int)(EE / 4));

  // 2) Q/K/V projections (bf16 out)
  dim3 gGrid(N / 256, M / 128);   // (4, 128)
  gemm_bf16_wmma<u16><<<gGrid, 256, 0, stream>>>(xb, wqb, bq, qb, M, N, K);
  gemm_bf16_wmma<u16><<<gGrid, 256, 0, stream>>>(xb, wkb, bk, kb, M, N, K);
  gemm_bf16_wmma<u16><<<gGrid, 256, 0, stream>>>(xb, wvb, bv, vb, M, N, K);

  // 3) per-token head-axis attention (overwrites xb with attention output)
  attn_headmix<<<M / 4, 128, 24576, stream>>>(qb, kb, vb, xb, is_causal);

  // 4) output projection (f32 out + bias)
  gemm_bf16_wmma<float><<<gGrid, 256, 0, stream>>>(xb, wob, bo, (float*)d_out, M, N, K);
}
